// Attention_17334488007302
// MI455X (gfx1250) — matrix operand order repaired and checked by reading, not compiled, not run on hardware
//
#include <hip/hip_runtime.h>

// -------- types --------
typedef __attribute__((ext_vector_type(16))) __bf16 v16bf;
typedef __attribute__((ext_vector_type(8)))  __bf16 v8bf;
typedef __attribute__((ext_vector_type(4)))  __bf16 v4bf;
typedef __attribute__((ext_vector_type(8)))  float  v8f;

#define OUT_ELEMS (8*1024*768)          // 6,291,456 floats (out), attn follows
#define HEAD_ELEMS (1024*64)            // per (b,h) Q/K/V elements

// -------- helpers --------
__device__ inline v8f zero_v8f() {
  v8f z;
#pragma unroll
  for (int i = 0; i < 8; ++i) z[i] = 0.0f;
  return z;
}
__device__ inline v8f wmma_bf16(v16bf a, v16bf b, v8f c) {
  // 8 args: (neg_a, A, neg_b, B, c_mod, C, reuse_a, reuse_b)
  return __builtin_amdgcn_wmma_f32_16x16x32_bf16(false, a, false, b, (short)0, c,
                                                 false, false);
}
__device__ inline v16bf concat8(v8bf a, v8bf b) {
  return __builtin_shufflevector(a, b, 0, 1, 2, 3, 4, 5, 6, 7,
                                 8, 9, 10, 11, 12, 13, 14, 15);
}

// A fragment 16x32 (MxK): lane l (0-15)/half h holds K = h*8+0..7 and 16+h*8+0..7
// memory order inside each 8-element group matches the fragment order -> pure loads
__device__ inline v16bf load_fragA_bf16(const unsigned short* base, int ld, int lane) {
  const int l = lane & 15, h = (lane >> 4) & 1;
  const __bf16* p = (const __bf16*)base + (size_t)l * ld + h * 8;
  v8bf lo = *(const v8bf*)(p);
  v8bf hi = *(const v8bf*)(p + 16);
  return concat8(lo, hi);
}
// B fragment 32x16 (KxN), source stored as rows-of-output-columns ([N,K] row-major):
// lane l = column n; half h holds K = h*8+0..7 and 16+h*8+0..7 (matches A)
__device__ inline v16bf load_fragB_bf16(const unsigned short* base, int ld, int lane) {
  const int l = lane & 15, h = (lane >> 4) & 1;
  const __bf16* p = (const __bf16*)base + (size_t)l * ld + h * 8;
  v8bf lo = *(const v8bf*)(p);
  v8bf hi = *(const v8bf*)(p + 16);
  return concat8(lo, hi);
}
// A fragment from f32 source (attn matrix), native cvt to bf16
__device__ inline v16bf load_fragA_f32(const float* base, int ld, int lane) {
  const int l = lane & 15, h = (lane >> 4) & 1;
  const float* p = base + (size_t)l * ld + h * 8;
  float4 f0 = *(const float4*)(p);
  float4 f1 = *(const float4*)(p + 4);
  float4 f2 = *(const float4*)(p + 16);
  float4 f3 = *(const float4*)(p + 20);
  v16bf r;
  r[0] = (__bf16)f0.x; r[1] = (__bf16)f0.y; r[2] = (__bf16)f0.z; r[3] = (__bf16)f0.w;
  r[4] = (__bf16)f1.x; r[5] = (__bf16)f1.y; r[6] = (__bf16)f1.z; r[7] = (__bf16)f1.w;
  r[8] = (__bf16)f2.x; r[9] = (__bf16)f2.y; r[10] = (__bf16)f2.z; r[11] = (__bf16)f2.w;
  r[12] = (__bf16)f3.x; r[13] = (__bf16)f3.y; r[14] = (__bf16)f3.z; r[15] = (__bf16)f3.w;
  return r;
}

// -------- kernel 0: elementwise f32 -> bf16 --------
__global__ void __launch_bounds__(256) cvt_f32_bf16(
    const float* __restrict__ in, unsigned short* __restrict__ outp, int n4) {
  const int i = blockIdx.x * 256 + threadIdx.x;
  if (i < n4) {
    float4 f = ((const float4*)in)[i];
    v4bf o;
    o[0] = (__bf16)f.x; o[1] = (__bf16)f.y; o[2] = (__bf16)f.z; o[3] = (__bf16)f.w;
    ((v4bf*)outp)[i] = o;
  }
}

// -------- kernel 1: QKV projection  Y[8192,2304] = Xbf[8192,768] . Wqkv^T --------
// epilogue scatters bf16 Q,K as [B*H, N, 64] and V transposed as [B*H, 64, N]
__global__ void __launch_bounds__(256) qkv_kernel(
    const unsigned short* __restrict__ xbf, const unsigned short* __restrict__ wqkv,
    unsigned short* __restrict__ qbf, unsigned short* __restrict__ kbf,
    unsigned short* __restrict__ vT) {
  const int lane = threadIdx.x & 31;
  const int wave = blockIdx.x * 8 + (threadIdx.x >> 5);
  const int tm = wave / 36, tn = wave % 36;
  const int m0 = tm * 32, n0 = tn * 64;

  v8f acc[2][4];
#pragma unroll
  for (int i = 0; i < 2; ++i)
#pragma unroll
    for (int t = 0; t < 4; ++t) acc[i][t] = zero_v8f();

  for (int kk = 0; kk < 768; kk += 32) {
    v16bf a0 = load_fragA_bf16(xbf + (size_t)m0 * 768 + kk, 768, lane);
    v16bf a1 = load_fragA_bf16(xbf + (size_t)(m0 + 16) * 768 + kk, 768, lane);
#pragma unroll
    for (int t = 0; t < 4; ++t) {
      v16bf b = load_fragB_bf16(wqkv + (size_t)(n0 + 16 * t) * 768 + kk, 768, lane);
      acc[0][t] = wmma_bf16(a0, b, acc[0][t]);
      acc[1][t] = wmma_bf16(a1, b, acc[1][t]);
    }
  }

  const int l = lane & 15, h = (lane >> 4) & 1;
  const int three = n0 / 768;          // 0=q, 1=k, 2=v  (tile never crosses)
  const int hh = (n0 % 768) / 64;      // head (tile never crosses a head)
  __bf16* qp = (__bf16*)qbf;
  __bf16* kp = (__bf16*)kbf;
  __bf16* vp = (__bf16*)vT;
#pragma unroll
  for (int i = 0; i < 2; ++i)
#pragma unroll
    for (int t = 0; t < 4; ++t) {
      const int d = t * 16 + l;        // 0..63 within head
#pragma unroll
      for (int r = 0; r < 8; ++r) {
        const int m = m0 + i * 16 + h * 8 + r;
        const int bb = m >> 10, nseq = m & 1023;
        const __bf16 bv = (__bf16)acc[i][t][r];
        const size_t head = (size_t)(bb * 12 + hh);
        if (three == 0)      qp[(head * 1024 + nseq) * 64 + d] = bv;
        else if (three == 1) kp[(head * 1024 + nseq) * 64 + d] = bv;
        else                 vp[(head * 64 + d) * 1024 + nseq] = bv;
      }
    }
}

// -------- kernel 2: scores S = scale * Q . K^T, f32 into attn region --------
__global__ void __launch_bounds__(256) scores_kernel(
    const unsigned short* __restrict__ qbf, const unsigned short* __restrict__ kbf,
    float* __restrict__ S) {
  const int lane = threadIdx.x & 31;
  const int bh = blockIdx.y;
  const int w = blockIdx.x * 8 + (threadIdx.x >> 5);   // 0..511
  const int m0 = (w >> 4) * 32, n0 = (w & 15) * 64;
  const unsigned short* Q = qbf + (size_t)bh * HEAD_ELEMS;
  const unsigned short* Km = kbf + (size_t)bh * HEAD_ELEMS;
  float* Sb = S + (size_t)bh * 1024 * 1024;

  v8f acc[2][4];
#pragma unroll
  for (int i = 0; i < 2; ++i)
#pragma unroll
    for (int t = 0; t < 4; ++t) acc[i][t] = zero_v8f();

#pragma unroll
  for (int kk = 0; kk < 64; kk += 32) {
    v16bf a0 = load_fragA_bf16(Q + (size_t)m0 * 64 + kk, 64, lane);
    v16bf a1 = load_fragA_bf16(Q + (size_t)(m0 + 16) * 64 + kk, 64, lane);
#pragma unroll
    for (int t = 0; t < 4; ++t) {
      v16bf b = load_fragB_bf16(Km + (size_t)(n0 + 16 * t) * 64 + kk, 64, lane);
      acc[0][t] = wmma_bf16(a0, b, acc[0][t]);
      acc[1][t] = wmma_bf16(a1, b, acc[1][t]);
    }
  }

  const int l = lane & 15, h = (lane >> 4) & 1;
#pragma unroll
  for (int i = 0; i < 2; ++i)
#pragma unroll
    for (int t = 0; t < 4; ++t)
#pragma unroll
      for (int r = 0; r < 8; ++r)
        Sb[(size_t)(m0 + i * 16 + h * 8 + r) * 1024 + n0 + t * 16 + l] =
            acc[i][t][r] * 0.125f;
}

// -------- kernel 3: row softmax over 1024, in place --------
__global__ void __launch_bounds__(256) softmax_kernel(float* __restrict__ attn) {
  __shared__ float red[256];
  float* p = attn + (size_t)blockIdx.x * 1024;
  const int tid = threadIdx.x;
  float v[4];
  float m = -3.0e38f;
#pragma unroll
  for (int j = 0; j < 4; ++j) { v[j] = p[tid + 256 * j]; m = fmaxf(m, v[j]); }
  red[tid] = m; __syncthreads();
  for (int s = 128; s > 0; s >>= 1) {
    if (tid < s) red[tid] = fmaxf(red[tid], red[tid + s]);
    __syncthreads();
  }
  m = red[0];
  __syncthreads();
  float sum = 0.0f;
#pragma unroll
  for (int j = 0; j < 4; ++j) { v[j] = __expf(v[j] - m); sum += v[j]; }
  red[tid] = sum; __syncthreads();
  for (int s = 128; s > 0; s >>= 1) {
    if (tid < s) red[tid] += red[tid + s];
    __syncthreads();
  }
  const float inv = 1.0f / red[0];
#pragma unroll
  for (int j = 0; j < 4; ++j) p[tid + 256 * j] = v[j] * inv;
}

// -------- kernel 4: O = attn(f32, cvt in loader) . V, bf16 out in [B,N,C] --------
__global__ void __launch_bounds__(256) av_kernel(
    const float* __restrict__ attn, const unsigned short* __restrict__ vT,
    unsigned short* __restrict__ obf) {
  const int lane = threadIdx.x & 31;
  const int bh = blockIdx.y;
  const int w = blockIdx.x * 8 + (threadIdx.x >> 5);   // 0..31
  const int m0 = w * 32;                               // n0 == 0 (N = 64)
  const float* A = attn + (size_t)bh * 1024 * 1024;
  const unsigned short* Vh = vT + (size_t)bh * HEAD_ELEMS;

  v8f acc[2][4];
#pragma unroll
  for (int i = 0; i < 2; ++i)
#pragma unroll
    for (int t = 0; t < 4; ++t) acc[i][t] = zero_v8f();

  for (int kk = 0; kk < 1024; kk += 32) {
    if (kk + 32 < 1024) __builtin_prefetch(A + (size_t)(m0 + lane) * 1024 + kk + 32, 0, 1);
    v16bf a0 = load_fragA_f32(A + (size_t)m0 * 1024 + kk, 1024, lane);
    v16bf a1 = load_fragA_f32(A + (size_t)(m0 + 16) * 1024 + kk, 1024, lane);
#pragma unroll
    for (int t = 0; t < 4; ++t) {
      v16bf b = load_fragB_bf16(Vh + (size_t)(16 * t) * 1024 + kk, 1024, lane);
      acc[0][t] = wmma_bf16(a0, b, acc[0][t]);
      acc[1][t] = wmma_bf16(a1, b, acc[1][t]);
    }
  }

  const int l = lane & 15, h = (lane >> 4) & 1;
  const int bq = bh / 12, hh = bh % 12;
  __bf16* op = (__bf16*)obf;
#pragma unroll
  for (int i = 0; i < 2; ++i)
#pragma unroll
    for (int t = 0; t < 4; ++t)
#pragma unroll
      for (int r = 0; r < 8; ++r) {
        const int nseq = m0 + i * 16 + h * 8 + r;
        const int d = t * 16 + l;
        op[((size_t)bq * 1024 + nseq) * 768 + hh * 64 + d] = (__bf16)acc[i][t][r];
      }
}

// -------- kernel 5: out = O . Wproj^T + b --------
__global__ void __launch_bounds__(256) proj_kernel(
    const unsigned short* __restrict__ obf, const unsigned short* __restrict__ wproj,
    const float* __restrict__ bproj, float* __restrict__ out) {
  const int lane = threadIdx.x & 31;
  const int w = blockIdx.x * 8 + (threadIdx.x >> 5);   // 0..3071
  const int m0 = (w / 12) * 32, n0 = (w % 12) * 64;

  v8f acc[2][4];
#pragma unroll
  for (int i = 0; i < 2; ++i)
#pragma unroll
    for (int t = 0; t < 4; ++t) acc[i][t] = zero_v8f();

  for (int kk = 0; kk < 768; kk += 32) {
    v16bf a0 = load_fragA_bf16(obf + (size_t)m0 * 768 + kk, 768, lane);
    v16bf a1 = load_fragA_bf16(obf + (size_t)(m0 + 16) * 768 + kk, 768, lane);
#pragma unroll
    for (int t = 0; t < 4; ++t) {
      v16bf b = load_fragB_bf16(wproj + (size_t)(n0 + 16 * t) * 768 + kk, 768, lane);
      acc[0][t] = wmma_bf16(a0, b, acc[0][t]);
      acc[1][t] = wmma_bf16(a1, b, acc[1][t]);
    }
  }

  const int l = lane & 15, h = (lane >> 4) & 1;
#pragma unroll
  for (int i = 0; i < 2; ++i)
#pragma unroll
    for (int t = 0; t < 4; ++t)
#pragma unroll
      for (int r = 0; r < 8; ++r) {
        const int row = m0 + i * 16 + h * 8 + r;
        const int col = n0 + t * 16 + l;
        out[(size_t)row * 768 + col] = acc[i][t][r] + bproj[col];
      }
}

extern "C" void kernel_launch(void* const* d_in, const int* in_sizes, int n_in,
                              void* d_out, int out_size, void* d_ws, size_t ws_size,
                              hipStream_t stream) {
  const float* x     = (const float*)d_in[0];
  const float* wqkv  = (const float*)d_in[1];
  const float* wproj = (const float*)d_in[2];
  const float* bproj = (const float*)d_in[3];

  float* out  = (float*)d_out;
  float* attn = out + OUT_ELEMS;       // [8,12,1024,1024] f32

  // workspace layout (bf16 elements, all 256B-aligned offsets): 67.6 MB total
  const size_t QKV_ELEMS = (size_t)8 * 12 * 1024 * 64;     // 6,291,456
  unsigned short* qbf   = (unsigned short*)d_ws;
  unsigned short* kbf   = qbf + QKV_ELEMS;
  unsigned short* vT    = kbf + QKV_ELEMS;
  unsigned short* obf   = vT + QKV_ELEMS;
  unsigned short* xbf   = obf + QKV_ELEMS;                 // 8192*768
  unsigned short* wqkvb = xbf + (size_t)8192 * 768;        // 2304*768
  unsigned short* wprjb = wqkvb + (size_t)2304 * 768;      // 768*768

  cvt_f32_bf16<<<6144, 256, 0, stream>>>(x, xbf, 8192 * 768 / 4);
  cvt_f32_bf16<<<1728, 256, 0, stream>>>(wqkv, wqkvb, 2304 * 768 / 4);
  cvt_f32_bf16<<<576, 256, 0, stream>>>(wproj, wprjb, 768 * 768 / 4);

  qkv_kernel<<<1152, 256, 0, stream>>>(xbf, wqkvb, qbf, kbf, vT);
  scores_kernel<<<dim3(64, 96), 256, 0, stream>>>(qbf, kbf, attn);
  softmax_kernel<<<98304, 256, 0, stream>>>(attn);
  av_kernel<<<dim3(4, 96), 256, 0, stream>>>(attn, vT, obf);
  proj_kernel<<<384, 256, 0, stream>>>(obf, wproj ? wprjb : wprjb, bproj, out);
}
